// ASR_Decoder_25855703122215
// MI455X (gfx1250) — matrix operand order repaired and checked by reading, not compiled, not run on hardware
//
#include <hip/hip_runtime.h>
#include <hip/hip_bf16.h>
#include <stdint.h>

// ---------------- problem constants ----------------
#define B_      32
#define TENC    1000
#define TDEC    200
#define LABEL_  5000
#define EMB_    512
#define ENC_    512
#define H1_     512
#define H2_     1024
#define ATT_    128
#define FN_     32
#define FL_     31
#define PAD_    15
#define KCAT    1536          // K for both gate GEMMs
#define NPROJ   5056          // 5000 padded to multiple of 64

typedef __bf16 bf16_t;
typedef __attribute__((ext_vector_type(16))) __bf16 v16bf;
typedef __attribute__((ext_vector_type(8)))  float  v8f;

__device__ __forceinline__ float sigm(float x) { return 1.0f / (1.0f + __expf(-x)); }

// ---------------- WMMA fragment loaders (CDNA5 16-bit layouts, ISA 7.12.2) ----
// A (16x32): lanes 0..15 = rows, K offsets {0..7,16..23}; lanes>=16 = {8..15,24..31}
__device__ __forceinline__ v16bf ldA(const bf16_t* A, int lda, int row, int k0, int lane) {
    const bf16_t* p = A + (size_t)row * lda + k0 + ((lane >> 4) << 3); // +0 or +8
    union { v16bf v; uint4 q[2]; } u;
    u.q[0] = *(const uint4*)(p);        // K offsets 0..7
    u.q[1] = *(const uint4*)(p + 16);   // K offsets 16..23
    return u.v;
}
// B (32x16 = K x N) from row-major W (N x K): same K order as A
// (lanes 0..15 -> K {0..7,16..23}, lanes 16..31 -> K {8..15,24..31})
__device__ __forceinline__ v16bf ldB(const bf16_t* W, int ldw, int col, int k0, int lane) {
    const bf16_t* p = W + (size_t)col * ldw + k0 + ((lane >> 4) << 3); // +0 or +8
    union { v16bf v; uint4 q[2]; } u;
    u.q[0] = *(const uint4*)(p);
    u.q[1] = *(const uint4*)(p + 16);
    return u.v;
}

// ---------------- generic bf16 WMMA GEMM:  C(MxN) = A(MxK) * W(NxK)^T + bias --
// One wave computes a 16x64 tile. Canonical software pipeline: prologue loads
// K-step 0; loop iterations load step s unconditionally while issuing the
// WMMAs of step s-1; epilogue does the last WMMA group.
// __launch_bounds__(128,1): let the allocator keep BOTH fragment sets
// (~2x40 VGPRs) live so the double buffer isn't collapsed into reloads.
__global__ void __launch_bounds__(128, 1)
gemm_bf16_wmma(const bf16_t* __restrict__ A, int lda,
               const bf16_t* __restrict__ W, int ldw,
               const float* __restrict__ bias,
               float* __restrict__ C, long long ldc,
               int Mtiles, int N64, int K, int nmax) {
    int lane = threadIdx.x & 31;
    int wid  = blockIdx.x * (blockDim.x >> 5) + (threadIdx.x >> 5);
    int total = Mtiles * N64;
    if (wid >= total) return;                 // wave-uniform: EXEC stays all-1s
    int mt = wid % Mtiles, nt = wid / Mtiles;
    int m0 = mt << 4, n0 = nt << 6;
    int arow  = m0 + (lane & 15);
    int bcol0 = n0 + (lane & 15);

    v8f c0 = {}, c1 = {}, c2 = {}, c3 = {};

    // prologue: fragments for K-step 0
    v16bf a_c  = ldA(A, lda, arow, 0, lane);
    v16bf b0_c = ldB(W, ldw, bcol0,      0, lane);
    v16bf b1_c = ldB(W, ldw, bcol0 + 16, 0, lane);
    v16bf b2_c = ldB(W, ldw, bcol0 + 32, 0, lane);
    v16bf b3_c = ldB(W, ldw, bcol0 + 48, 0, lane);

    int nsteps = K >> 5;                      // K is a multiple of 32
    for (int s = 1; s < nsteps; ++s) {
        int k1 = s << 5;
        // distant prefetch of the weight stream (gfx1250 global_prefetch path);
        // worst case touches the adjacent workspace region -> still valid.
        __builtin_prefetch(W + (size_t)bcol0 * ldw + k1 + 32, 0, 1);
        v16bf a_n  = ldA(A, lda, arow, k1, lane);
        v16bf b0_n = ldB(W, ldw, bcol0,      k1, lane);
        v16bf b1_n = ldB(W, ldw, bcol0 + 16, k1, lane);
        v16bf b2_n = ldB(W, ldw, bcol0 + 32, k1, lane);
        v16bf b3_n = ldB(W, ldw, bcol0 + 48, k1, lane);
        c0 = __builtin_amdgcn_wmma_f32_16x16x32_bf16(false, a_c, false, b0_c, (short)0, c0, false, false);
        c1 = __builtin_amdgcn_wmma_f32_16x16x32_bf16(false, a_c, false, b1_c, (short)0, c1, false, false);
        c2 = __builtin_amdgcn_wmma_f32_16x16x32_bf16(false, a_c, false, b2_c, (short)0, c2, false, false);
        c3 = __builtin_amdgcn_wmma_f32_16x16x32_bf16(false, a_c, false, b3_c, (short)0, c3, false, false);
        a_c = a_n; b0_c = b0_n; b1_c = b1_n; b2_c = b2_n; b3_c = b3_n;
    }
    // epilogue: last K-step
    c0 = __builtin_amdgcn_wmma_f32_16x16x32_bf16(false, a_c, false, b0_c, (short)0, c0, false, false);
    c1 = __builtin_amdgcn_wmma_f32_16x16x32_bf16(false, a_c, false, b1_c, (short)0, c1, false, false);
    c2 = __builtin_amdgcn_wmma_f32_16x16x32_bf16(false, a_c, false, b2_c, (short)0, c2, false, false);
    c3 = __builtin_amdgcn_wmma_f32_16x16x32_bf16(false, a_c, false, b3_c, (short)0, c3, false, false);

    // C/D layout: lane%16 = N col; VGPR r -> M = r (lanes<16) / 8+r (lanes>=16)
    int nl = lane & 15;
    int mb = m0 + ((lane >> 4) << 3);
#pragma unroll
    for (int tn = 0; tn < 4; ++tn) {
        int n = n0 + tn * 16 + nl;
        if (n < nmax) {
            float bb = bias ? bias[n] : 0.0f;
            v8f cc = (tn == 0) ? c0 : ((tn == 1) ? c1 : ((tn == 2) ? c2 : c3));
#pragma unroll
            for (int r = 0; r < 8; ++r)
                C[(size_t)(mb + r) * (size_t)ldc + n] = cc[r] + bb;
        }
    }
}

// ---------------- prep kernels ----------------
__global__ void cvt_plain(bf16_t* __restrict__ dst, const float* __restrict__ src, int n) {
    int i = blockIdx.x * blockDim.x + threadIdx.x;
    if (i < n) dst[i] = (bf16_t)src[i];
}
__global__ void cvt_cat2(bf16_t* __restrict__ dst, const float* __restrict__ A, int wa,
                         const float* __restrict__ Bm, int wb, int rows) {
    int w = wa + wb;
    int i = blockIdx.x * blockDim.x + threadIdx.x;
    if (i >= rows * w) return;
    int r = i / w, c = i % w;
    float v = (c < wa) ? A[(size_t)r * wa + c] : Bm[(size_t)r * wb + (c - wa)];
    dst[i] = (bf16_t)v;
}
__global__ void cvt_pad(bf16_t* __restrict__ dst, const float* __restrict__ src,
                        int rows_src, int rows_tot, int cols) {
    int i = blockIdx.x * blockDim.x + threadIdx.x;
    if (i >= rows_tot * cols) return;
    int r = i / cols;
    dst[i] = (bf16_t)((r < rows_src) ? src[i] : 0.0f);
}
__global__ void addvec(float* __restrict__ o, const float* __restrict__ a,
                       const float* __restrict__ b, int n) {
    int i = blockIdx.x * blockDim.x + threadIdx.x;
    if (i < n) o[i] = a[i] + b[i];
}

// ---------------- per-step kernels ----------------
__global__ void gather_emb(bf16_t* __restrict__ xcat1, const int* __restrict__ dec,
                           const float* __restrict__ emb, int t) {
    int i = blockIdx.x * blockDim.x + threadIdx.x;   // 32*512
    if (i >= B_ * EMB_) return;
    int b = i >> 9, j = i & 511;
    int tok = (t == 0) ? 1 : dec[b * TDEC + (t - 1)];
    xcat1[(size_t)b * KCAT + j] = (bf16_t)emb[(size_t)tok * EMB_ + j];
}

__global__ void lstm_pw(const float* __restrict__ G, float* __restrict__ c,
                        bf16_t* __restrict__ hdst, int H, int hstride) {
    int i = blockIdx.x * blockDim.x + threadIdx.x;
    if (i >= B_ * H) return;
    int b = i / H, j = i % H;
    const float* g = G + (size_t)b * 4 * H;
    float gi = g[j], gf = g[H + j], gg = g[2 * H + j], go = g[3 * H + j];
    float cn = sigm(gf) * c[i] + sigm(gi) * tanhf(gg);
    c[i] = cn;
    hdst[(size_t)b * hstride + j] = (bf16_t)(sigm(go) * tanhf(cn));
}

// Fused location-sensitive attention: conv -> energies -> softmax (in place on
// w/wcum) -> context matvec. One block per batch row, 256 threads.
// Safe in-place: all reads of previous w/wcum finish before the second barrier.
__global__ void __launch_bounds__(256) attention_k(
        float* __restrict__ w, float* __restrict__ wcum,
        const float* __restrict__ q, const float* __restrict__ pm,
        const float* __restrict__ conv, const float* __restrict__ locW,
        const float* __restrict__ vv, const float* __restrict__ enc,
        bf16_t* __restrict__ xcat1, bf16_t* __restrict__ xcat2) {
    __shared__ float sconv[FN_ * 2 * FL_];   // conv[f][c][j]
    __shared__ float slw[ATT_ * FN_];
    __shared__ float sv[ATT_];
    __shared__ float sq[ATT_];
    __shared__ float red[256];
    __shared__ float se[TENC];               // energies, then normalized weights
    int b = blockIdx.x, tid = threadIdx.x;
    for (int i = tid; i < FN_ * 2 * FL_; i += 256) sconv[i] = conv[i];
    for (int i = tid; i < ATT_ * FN_;   i += 256) slw[i]   = locW[i];
    if (tid < ATT_) { sv[tid] = vv[tid]; sq[tid] = q[b * ATT_ + tid]; }
    __syncthreads();

    const float* wp  = w    + b * TENC;
    const float* wcp = wcum + b * TENC;
    for (int t = tid; t < TENC; t += 256) {
        float f[FN_];
#pragma unroll
        for (int ff = 0; ff < FN_; ++ff) f[ff] = 0.0f;
        for (int j = 0; j < FL_; ++j) {
            int ti = t + j - PAD_;
            if ((unsigned)ti < (unsigned)TENC) {
                float wv = wp[ti], wcv = wcp[ti];
#pragma unroll
                for (int ff = 0; ff < FN_; ++ff)
                    f[ff] += wv * sconv[ff * 62 + j] + wcv * sconv[ff * 62 + FL_ + j];
            }
        }
        const float* pmr = pm + ((size_t)b * TENC + t) * ATT_;
        float acc = 0.0f;
        for (int a = 0; a < ATT_; ++a) {
            float la = 0.0f;
            const float* lr = slw + a * FN_;
#pragma unroll
            for (int ff = 0; ff < FN_; ++ff) la += lr[ff] * f[ff];
            acc += sv[a] * tanhf(sq[a] + la + pmr[a]);
        }
        se[t] = acc;
    }
    __syncthreads();     // all reads of previous w/wcum complete beyond here

    // softmax over se[0..999]
    float m = -1e30f;
    for (int t = tid; t < TENC; t += 256) m = fmaxf(m, se[t]);
    red[tid] = m; __syncthreads();
    for (int s = 128; s > 0; s >>= 1) { if (tid < s) red[tid] = fmaxf(red[tid], red[tid + s]); __syncthreads(); }
    m = red[0]; __syncthreads();
    float lsum = 0.0f;
    for (int t = tid; t < TENC; t += 256) lsum += __expf(se[t] - m);
    red[tid] = lsum; __syncthreads();
    for (int s = 128; s > 0; s >>= 1) { if (tid < s) red[tid] += red[tid + s]; __syncthreads(); }
    float inv = 1.0f / red[0];
    __syncthreads();
    for (int t = tid; t < TENC; t += 256) {
        float wv = __expf(se[t] - m) * inv;
        se[t] = wv;                       // normalized weight for context
        w[b * TENC + t]    = wv;          // persist for next step's conv
        wcum[b * TENC + t] += wv;
    }
    __syncthreads();

    // ctx[b,e] = sum_t w[b,t] * enc[b,t,e]; bf16 into both xcat slots
    for (int e = tid; e < ENC_; e += 256) {
        const float* eb = enc + (size_t)b * TENC * ENC_ + e;
        float acc = 0.0f;
        for (int t = 0; t < TENC; ++t) acc += se[t] * eb[(size_t)t * ENC_];
        bf16_t cv = (bf16_t)acc;
        xcat1[(size_t)b * KCAT + EMB_ + e] = cv;  // ctx slot for next gates1
        xcat2[(size_t)b * KCAT + e]        = cv;  // ctx slot for gates2
    }
}

// ---------------- host side ----------------
extern "C" void kernel_launch(void* const* d_in, const int* in_sizes, int n_in,
                              void* d_out, int out_size, void* d_ws, size_t ws_size,
                              hipStream_t stream) {
    const float* enc      = (const float*)d_in[0];
    const int*   dec      = (const int*)  d_in[1];
    const float* emb      = (const float*)d_in[2];
    const float* att_Wih  = (const float*)d_in[3];
    const float* att_Whh  = (const float*)d_in[4];
    const float* att_bih  = (const float*)d_in[5];
    const float* att_bhh  = (const float*)d_in[6];
    const float* dec_Wih  = (const float*)d_in[7];
    const float* dec_Whh  = (const float*)d_in[8];
    const float* dec_bih  = (const float*)d_in[9];
    const float* dec_bhh  = (const float*)d_in[10];
    const float* mem_W    = (const float*)d_in[11];
    const float* mem_b    = (const float*)d_in[12];
    const float* query_W  = (const float*)d_in[13];
    const float* loc_conv = (const float*)d_in[14];
    const float* loc_W    = (const float*)d_in[15];
    const float* vvec     = (const float*)d_in[16];
    const float* proj_W   = (const float*)d_in[17];
    float* out = (float*)d_out;

    // ---- carve workspace (all offsets 256B aligned; total ~81 MB) ----
    size_t off = 0;
    char* base = (char*)d_ws;
    auto alloc = [&](size_t bytes) -> void* {
        void* p = base + off;
        off = (off + bytes + 255) & ~(size_t)255;
        return p;
    };
    bf16_t* encbf = (bf16_t*)alloc((size_t)B_ * TENC * ENC_ * 2);
    float*  pm    = (float*) alloc((size_t)B_ * TENC * ATT_ * 4);
    bf16_t* w1    = (bf16_t*)alloc((size_t)4 * H1_ * KCAT * 2);
    bf16_t* w2    = (bf16_t*)alloc((size_t)4 * H2_ * KCAT * 2);
    bf16_t* wq    = (bf16_t*)alloc((size_t)ATT_ * H1_ * 2);
    bf16_t* wmem  = (bf16_t*)alloc((size_t)ATT_ * ENC_ * 2);
    bf16_t* wproj = (bf16_t*)alloc((size_t)NPROJ * H2_ * 2);
    float*  bias1 = (float*) alloc((size_t)4 * H1_ * 4);
    float*  bias2 = (float*) alloc((size_t)4 * H2_ * 4);
    bf16_t* xcat1 = (bf16_t*)alloc((size_t)B_ * KCAT * 2);   // [emb | ctx | h1]
    bf16_t* xcat2 = (bf16_t*)alloc((size_t)B_ * KCAT * 2);   // [ctx | h2]
    float*  G1    = (float*) alloc((size_t)B_ * 4 * H1_ * 4);
    float*  G2    = (float*) alloc((size_t)B_ * 4 * H2_ * 4);
    float*  c1    = (float*) alloc((size_t)B_ * H1_ * 4);
    float*  c2    = (float*) alloc((size_t)B_ * H2_ * 4);
    float*  qbuf  = (float*) alloc((size_t)B_ * ATT_ * 4);
    float*  wbuf  = (float*) alloc((size_t)B_ * TENC * 4);
    float*  wcum  = (float*) alloc((size_t)B_ * TENC * 4);

    // ---- zero recurrent state (graph-capture-legal) ----
    hipMemsetAsync(xcat1, 0, (size_t)B_ * KCAT * 2, stream);
    hipMemsetAsync(xcat2, 0, (size_t)B_ * KCAT * 2, stream);
    hipMemsetAsync(c1,    0, (size_t)B_ * H1_ * 4, stream);
    hipMemsetAsync(c2,    0, (size_t)B_ * H2_ * 4, stream);
    hipMemsetAsync(wbuf,  0, (size_t)B_ * TENC * 4, stream);
    hipMemsetAsync(wcum,  0, (size_t)B_ * TENC * 4, stream);

    // ---- one-time prep: bf16 casts + fused weight layouts ----
    {
        int n = B_ * TENC * ENC_;
        cvt_plain<<<(n + 255) / 256, 256, 0, stream>>>(encbf, enc, n);
        n = ATT_ * H1_;
        cvt_plain<<<(n + 255) / 256, 256, 0, stream>>>(wq, query_W, n);
        n = ATT_ * ENC_;
        cvt_plain<<<(n + 255) / 256, 256, 0, stream>>>(wmem, mem_W, n);
        n = 4 * H1_ * KCAT;
        cvt_cat2<<<(n + 255) / 256, 256, 0, stream>>>(w1, att_Wih, EMB_ + H1_, att_Whh, H1_, 4 * H1_);
        n = 4 * H2_ * KCAT;
        cvt_cat2<<<(n + 255) / 256, 256, 0, stream>>>(w2, dec_Wih, H1_, dec_Whh, H2_, 4 * H2_);
        n = NPROJ * H2_;
        cvt_pad<<<(n + 255) / 256, 256, 0, stream>>>(wproj, proj_W, LABEL_, NPROJ, H2_);
        addvec<<<(4 * H1_ + 255) / 256, 256, 0, stream>>>(bias1, att_bih, att_bhh, 4 * H1_);
        addvec<<<(4 * H2_ + 255) / 256, 256, 0, stream>>>(bias2, dec_bih, dec_bhh, 4 * H2_);
    }

    // ---- processed_memory = enc @ mem_W^T + mem_b : (32000 x 128), K=512 ----
    {
        int Mtiles = (B_ * TENC) / 16, N64 = ATT_ / 64;      // 2000 x 2
        int waves = Mtiles * N64;
        gemm_bf16_wmma<<<(waves + 3) / 4, 128, 0, stream>>>(
            encbf, ENC_, wmem, ENC_, mem_b, pm, (long long)ATT_, Mtiles, N64, ENC_, ATT_);
    }

    // ---- 200 sequential decode steps (8 launches/step) ----
    for (int t = 0; t < TDEC; ++t) {
        gather_emb<<<(B_ * EMB_ + 255) / 256, 256, 0, stream>>>(xcat1, dec, emb, t);

        // attention-LSTM gates: (32 x 2048) = xcat1(32x1536) @ w1^T
        { int w = 2 * (4 * H1_ / 64);
          gemm_bf16_wmma<<<(w + 3) / 4, 128, 0, stream>>>(
              xcat1, KCAT, w1, KCAT, bias1, G1, (long long)(4 * H1_), 2, 4 * H1_ / 64, KCAT, 4 * H1_); }
        lstm_pw<<<(B_ * H1_ + 255) / 256, 256, 0, stream>>>(G1, c1, xcat1 + EMB_ + H1_, H1_, KCAT);

        // q = h1 @ query_W^T : (32 x 128), A = h1 slot of xcat1 (lda=1536)
        { int w = 2 * (ATT_ / 64);
          gemm_bf16_wmma<<<(w + 3) / 4, 128, 0, stream>>>(
              xcat1 + EMB_ + H1_, KCAT, wq, H1_, nullptr, qbuf, (long long)ATT_, 2, ATT_ / 64, H1_, ATT_); }

        // fused conv + energies + softmax + context
        attention_k<<<B_, 256, 0, stream>>>(wbuf, wcum, qbuf, pm, loc_conv,
                                            loc_W, vvec, enc, xcat1, xcat2);

        // decoder-LSTM gates: (32 x 4096) = xcat2(32x1536) @ w2^T
        { int w = 2 * (4 * H2_ / 64);
          gemm_bf16_wmma<<<(w + 3) / 4, 128, 0, stream>>>(
              xcat2, KCAT, w2, KCAT, bias2, G2, (long long)(4 * H2_), 2, 4 * H2_ / 64, KCAT, 4 * H2_); }
        lstm_pw<<<(B_ * H2_ + 255) / 256, 256, 0, stream>>>(G2, c2, xcat2 + H1_, H2_, KCAT);

        // logits: (32 x 5000) = h2 @ proj_W^T, strided store into out[b][t][:]
        { int w = 2 * (NPROJ / 64);
          gemm_bf16_wmma<<<(w + 3) / 4, 128, 0, stream>>>(
              xcat2 + H1_, KCAT, wproj, H2_, nullptr,
              out + (size_t)t * LABEL_, (long long)TDEC * LABEL_, 2, NPROJ / 64, H2_, LABEL_); }
    }
}
